// PerceiverAttention_modified_75222057222589
// MI455X (gfx1250) — matrix operand order repaired and checked by reading, not compiled, not run on hardware
//
#include <hip/hip_runtime.h>

// ---------------------------------------------------------------------------
// PerceiverAttention forward for MI455X (gfx1250, wave32, WMMA).
// All GEMMs in NT form (C = A * B^T, contraction innermost in both operands)
// so A/B WMMA fragments are contiguous b128 loads per the CDNA5 16-bit
// A/B VGPR layouts. bf16 operands, f32 accumulate via v_wmma_f32_16x16x32_bf16.
// K-loop is two-phase register double-buffered so WMMAs consume fragments
// loaded one phase earlier (no s_wait_loadcnt 0x0 stall per WMMA).
// Wave tile 32x64: 8 WMMAs per phase reuse 4 B fragments across 2 A fragments.
// ---------------------------------------------------------------------------

typedef unsigned short u16;
typedef unsigned int   u32;
typedef __attribute__((ext_vector_type(16))) __bf16 v16bf;
typedef __attribute__((ext_vector_type(8)))  float  v8f;
typedef __attribute__((ext_vector_type(4)))  u32    u32x4;

#define BATCH 8
#define SEQN  4096
#define SEQM  64
#define DIM   1024
#define HEADS 16
#define DHEAD 64
#define INNER 1024          // HEADS*DHEAD
#define ZB    (BATCH*HEADS) // 128

union FragAB { v16bf v; u32x4 u[2]; };

__device__ __forceinline__ u16 f32_to_bf16(float f) {
  u32 x = __builtin_bit_cast(u32, f);
  x += 0x7FFFu + ((x >> 16) & 1u);        // round-to-nearest-even
  return (u16)(x >> 16);
}

__device__ __forceinline__ void storeC(float* p, float v) { *p = v; }
__device__ __forceinline__ void storeC(u16*   p, float v) { *p = f32_to_bf16(v); }

// --------------------------- elementwise casts -----------------------------

__global__ void cvt_f32_bf16(const float* __restrict__ s, u16* __restrict__ d,
                             long long count) {
  long long i = (long long)blockIdx.x * blockDim.x + threadIdx.x;
  long long st = (long long)gridDim.x * blockDim.x;
  for (; i < count; i += st) d[i] = f32_to_bf16(s[i]);
}

// --------------------------- generic NT WMMA GEMM --------------------------

__device__ __forceinline__ void load_a2(FragAB a[2], const u16* r0,
                                        const u16* r1, int k0) {
  a[0].u[0] = *(const u32x4*)(r0 + k0);
  a[0].u[1] = *(const u32x4*)(r0 + k0 + 16);
  a[1].u[0] = *(const u32x4*)(r1 + k0);
  a[1].u[1] = *(const u32x4*)(r1 + k0 + 16);
}

__device__ __forceinline__ void load_b4(FragAB b[4], const u16* br0,
                                        long long bstep, int k0) {
#pragma unroll
  for (int t = 0; t < 4; ++t) {
    const u16* bp = br0 + bstep * t + k0;
    b[t].u[0] = *(const u32x4*)(bp);
    b[t].u[1] = *(const u32x4*)(bp + 16);
  }
}

__device__ __forceinline__ void mma8(v8f acc[2][4], const FragAB a[2],
                                     const FragAB b[4]) {
#pragma unroll
  for (int i = 0; i < 2; ++i)
#pragma unroll
    for (int t = 0; t < 4; ++t)
      acc[i][t] = __builtin_amdgcn_wmma_f32_16x16x32_bf16(
          false, a[i].v, false, b[t].v, (short)0, acc[i][t], false, false);
}

// C[z0*cs0 + z1*cs1 + row*ldc + col] = scale * sum_k A[z*as + row*lda + k] *
//                                                   B[z*bs + col*ldb + k]
// Block: 256 threads = 8 waves arranged 2(M)x4(N); wave tile 32x64.
// Requires M%64==0 (or wave skip), N%64==0, K%64==0 (true for all call sites).

template <typename CT>
__global__ __launch_bounds__(256) void gemm_nt_wmma_bf16(
    const u16* __restrict__ A, const u16* __restrict__ B, CT* __restrict__ C,
    int M, int N, int K, int lda, int ldb, int ldc,
    long long as, long long bs, long long cs0, long long cs1, int zdiv,
    float scale)
{
  const int z  = blockIdx.z;
  const int z0 = z / zdiv, z1 = z % zdiv;
  const u16* Ab = A + (long long)z * as;
  const u16* Bb = B + (long long)z * bs;
  CT* Cb = C + (long long)z0 * cs0 + (long long)z1 * cs1;

  const int w    = threadIdx.x >> 5;
  const int lane = threadIdx.x & 31;
  const int m0 = blockIdx.y * 64  + (w >> 2) * 32;
  const int n0 = blockIdx.x * 256 + (w & 3) * 64;
  if (m0 >= M || n0 >= N) return;   // wave-uniform -> EXEC all-ones inside

  v8f acc[2][4];
#pragma unroll
  for (int i = 0; i < 2; ++i)
#pragma unroll
    for (int t = 0; t < 4; ++t)
#pragma unroll
      for (int e = 0; e < 8; ++e) acc[i][t][e] = 0.0f;

  const int hsel = lane >> 4;       // K-half select per ISA 16-bit layout
  const int l15  = lane & 15;
  // A fragment rows m0+l15 and m0+16+l15: chunks [k0+8h..+8) and [+16..+8)
  const u16* ar0 = Ab + (long long)(m0 + l15) * lda + 8 * hsel;
  const u16* ar1 = ar0 + (long long)16 * lda;
  // B fragment row n0+16t+l15: chunks [k0+8h..+8) and [+16..+8)
  const u16* br0 = Bb + (long long)(n0 + l15) * ldb + 8 * hsel;
  const long long bstep = (long long)16 * ldb;

  // Two-phase register double buffering over K (K % 64 == 0, K >= 64).
  FragAB a0[2], a1[2], b0[4], b1[4];
  load_a2(a0, ar0, ar1, 0);
  load_b4(b0, br0, bstep, 0);
  for (int k0 = 0; k0 < K; k0 += 64) {
    const int k1 = k0 + 32, k2 = k0 + 64;
    load_a2(a1, ar0, ar1, k1);      // phase-1 loads in flight ...
    load_b4(b1, br0, bstep, k1);
    mma8(acc, a0, b0);              // ... while phase-0 WMMAs execute
    if (k2 < K) {
      load_a2(a0, ar0, ar1, k2);    // phase-0 loads for next iter in flight
      load_b4(b0, br0, bstep, k2);
    }
    mma8(acc, a1, b1);
  }

  // C/D layout: VGPR v, lanes 0-15: (M=v, N=lane); lanes 16-31: (M=v+8).
#pragma unroll
  for (int i = 0; i < 2; ++i)
#pragma unroll
    for (int t = 0; t < 4; ++t) {
      const int col = n0 + 16 * t + l15;
#pragma unroll
      for (int v = 0; v < 8; ++v) {
        const int row = m0 + 16 * i + v + 8 * hsel;
        storeC(Cb + (long long)row * ldc + col, acc[i][t][v] * scale);
      }
    }
}

// --------------------------- reshapes --------------------------------------

// kv_raw [b*n, 2*inner] -> k [b,h,n,d] and vT [b,h,d,n]
__global__ void reshape_kv(const u16* __restrict__ kvb, u16* __restrict__ kh,
                           u16* __restrict__ vT) {
  const long long total = (long long)BATCH * SEQN * 2 * INNER;
  long long i = (long long)blockIdx.x * blockDim.x + threadIdx.x;
  long long st = (long long)gridDim.x * blockDim.x;
  for (; i < total; i += st) {
    long long r = i >> 11;             // / 2048
    int c = (int)(i & 2047);
    int b = (int)(r >> 12);            // / 4096
    int n = (int)(r & 4095);
    u16 val = kvb[i];
    if (c < INNER) {
      int h = c >> 6, d = c & 63;
      kh[(((long long)(b * HEADS + h) * SEQN + n) << 6) + d] = val;
    } else {
      int c2 = c - INNER, h = c2 >> 6, d = c2 & 63;
      vT[((long long)(b * HEADS + h) * DHEAD + d) * SEQN + n] = val;
    }
  }
}

// q_raw [b*m, inner] -> qh [b,h,m,d]
__global__ void reshape_q(const u16* __restrict__ qb, u16* __restrict__ qh) {
  const long long total = (long long)BATCH * SEQM * INNER;
  long long i = (long long)blockIdx.x * blockDim.x + threadIdx.x;
  long long st = (long long)gridDim.x * blockDim.x;
  for (; i < total; i += st) {
    long long r = i >> 10;
    int c = (int)(i & 1023);
    int b = (int)(r >> 6), m = (int)(r & 63);
    int h = c >> 6, d = c & 63;
    qh[(((long long)(b * HEADS + h) * SEQM + m) << 6) + d] = qb[i];
  }
}

// --------------------------- softmax / loss --------------------------------

// Row max over unmasked j for each (b,h,m) row of sim.
__global__ __launch_bounds__(256) void rowmax_kernel(
    const float* __restrict__ sim, const unsigned char* __restrict__ mask,
    float* __restrict__ rmax) {
  const int row = blockIdx.x;          // z*64 + m, z = b*16 + h
  const int b = row >> 10;             // / (HEADS*SEQM) = /1024
  const float* srow = sim + (long long)row * SEQN;
  const unsigned char* mrow = mask + (long long)b * SEQN;
  float mx = -3.402823466e38f;
  for (int j = threadIdx.x; j < SEQN; j += 256)
    if (!mrow[j]) mx = fmaxf(mx, srow[j]);
  __shared__ float red[256];
  red[threadIdx.x] = mx;
  __syncthreads();
  for (int s = 128; s > 0; s >>= 1) {
    if (threadIdx.x < s) red[threadIdx.x] = fmaxf(red[threadIdx.x], red[threadIdx.x + s]);
    __syncthreads();
  }
  if (threadIdx.x == 0) rmax[row] = red[0];
}

// Unbiased cross-query variance per (b,h,n); relu(1 - std) summed into acc.
// Masked columns contribute exactly 0 (std overridden to 1 in the reference).
__global__ __launch_bounds__(256) void colvar_loss_kernel(
    const float* __restrict__ sim, const unsigned char* __restrict__ mask,
    const float* __restrict__ rmax, float* __restrict__ acc) {
  const int z = blockIdx.y;            // b*16 + h
  const int n = blockIdx.x * 256 + threadIdx.x;
  const int b = z >> 4;
  __shared__ float rm[SEQM];
  if (threadIdx.x < SEQM) rm[threadIdx.x] = rmax[z * SEQM + threadIdx.x];
  __syncthreads();
  float contrib = 0.0f;
  if (!mask[(long long)b * SEQN + n]) {
    float s = 0.0f, ss = 0.0f;
    const float* base = sim + (long long)z * SEQM * SEQN + n;
    for (int m = 0; m < SEQM; ++m) {
      float x = base[(long long)m * SEQN] - rm[m];
      s += x; ss += x * x;
    }
    float var = (ss - s * s * (1.0f / SEQM)) * (1.0f / (SEQM - 1));
    float sd  = sqrtf(var + 1e-4f);
    contrib = fmaxf(0.0f, 1.0f - sd);
  }
  __shared__ float red[256];
  red[threadIdx.x] = contrib;
  __syncthreads();
  for (int s = 128; s > 0; s >>= 1) {
    if (threadIdx.x < s) red[threadIdx.x] += red[threadIdx.x + s];
    __syncthreads();
  }
  if (threadIdx.x == 0) atomicAdd(acc, red[0]);
}

// Masked softmax row -> bf16 attn probabilities.
__global__ __launch_bounds__(256) void softmax_kernel(
    const float* __restrict__ sim, const unsigned char* __restrict__ mask,
    const float* __restrict__ rmax, u16* __restrict__ attn) {
  const int row = blockIdx.x;
  const int b = row >> 10;
  const float* srow = sim + (long long)row * SEQN;
  const unsigned char* mrow = mask + (long long)b * SEQN;
  const float rm = rmax[row];
  float s = 0.0f;
  for (int j = threadIdx.x; j < SEQN; j += 256)
    if (!mrow[j]) s += __expf(srow[j] - rm);
  __shared__ float red[256];
  red[threadIdx.x] = s;
  __syncthreads();
  for (int st = 128; st > 0; st >>= 1) {
    if (threadIdx.x < st) red[threadIdx.x] += red[threadIdx.x + st];
    __syncthreads();
  }
  const float inv = (red[0] > 0.0f) ? (1.0f / red[0]) : 0.0f;
  u16* arow = attn + (long long)row * SEQN;
  for (int j = threadIdx.x; j < SEQN; j += 256) {
    float p = mrow[j] ? 0.0f : __expf(srow[j] - rm) * inv;
    arow[j] = f32_to_bf16(p);
  }
}

__global__ void zero1_kernel(float* p) { *p = 0.0f; }

__global__ void loss_write_kernel(const float* __restrict__ acc,
                                  float* __restrict__ out) {
  out[0] = acc[0] * (1.0f / (float)((long long)BATCH * HEADS * SEQN));
}

// --------------------------- orchestration ---------------------------------

extern "C" void kernel_launch(void* const* d_in, const int* in_sizes, int n_in,
                              void* d_out, int out_size, void* d_ws, size_t ws_size,
                              hipStream_t stream) {
  const float* x    = (const float*)d_in[0];   // [8,4096,1024]
  const float* lat  = (const float*)d_in[1];   // [8,64,1024]
  const unsigned char* mask = (const unsigned char*)d_in[2]; // bool [8,4096]
  const float* Wq   = (const float*)d_in[3];   // [1024,1024]
  const float* Wkv  = (const float*)d_in[4];   // [2048,1024]
  const float* Wout = (const float*)d_in[5];   // [1024,1024]
  float* out = (float*)d_out;                  // [8,64,1024] ++ [1] loss

  // ---- workspace layout (bytes, 256-aligned). Total ~349 MB. ----
  char* ws = (char*)d_ws;
  size_t off = 0;
  auto take = [&](size_t bytes) { char* p = ws + off; off += (bytes + 255) & ~(size_t)255; return p; };
  u16*   xb    = (u16*)take((size_t)BATCH * SEQN * DIM * 2);       // 67 MB (aliased by attn later)
  u16*   latb  = (u16*)take((size_t)BATCH * SEQM * DIM * 2);       // 1 MB
  u16*   wqb   = (u16*)take((size_t)INNER * DIM * 2);              // 2 MB
  u16*   wkvb  = (u16*)take((size_t)2 * INNER * DIM * 2);          // 4 MB
  u16*   woutb = (u16*)take((size_t)DIM * INNER * 2);              // 2 MB
  u16*   kvb   = (u16*)take((size_t)BATCH * SEQN * 2 * INNER * 2); // 134 MB (aliased by sim later)
  u16*   qb    = (u16*)take((size_t)BATCH * SEQM * INNER * 2);     // 1 MB
  u16*   qh    = (u16*)take((size_t)BATCH * SEQM * INNER * 2);     // 1 MB
  u16*   kh    = (u16*)take((size_t)BATCH * SEQN * INNER * 2);     // 67 MB
  u16*   vT    = (u16*)take((size_t)BATCH * SEQN * INNER * 2);     // 67 MB
  u16*   outh  = (u16*)take((size_t)BATCH * SEQM * INNER * 2);     // 1 MB
  float* rmax  = (float*)take((size_t)ZB * SEQM * 4);              // 32 KB
  float* lacc  = (float*)take(256);
  // Aliases: sim (f32, 134 MB) reuses kvb (dead after reshape_kv);
  //          attn (bf16, 67 MB) reuses xb (dead after the KV GEMM).
  float* sim  = (float*)kvb;
  u16*   attn = (u16*)xb;
  (void)ws_size; (void)in_sizes; (void)n_in; (void)out_size;

  zero1_kernel<<<1, 1, 0, stream>>>(lacc);

  // f32 -> bf16 casts
  cvt_f32_bf16<<<2048, 256, 0, stream>>>(x,    xb,    (long long)BATCH * SEQN * DIM);
  cvt_f32_bf16<<<256,  256, 0, stream>>>(lat,  latb,  (long long)BATCH * SEQM * DIM);
  cvt_f32_bf16<<<512,  256, 0, stream>>>(Wq,   wqb,   (long long)INNER * DIM);
  cvt_f32_bf16<<<1024, 256, 0, stream>>>(Wkv,  wkvb,  (long long)2 * INNER * DIM);
  cvt_f32_bf16<<<512,  256, 0, stream>>>(Wout, woutb, (long long)DIM * INNER);

  // kv = x @ Wkv^T : M=32768 N=2048 K=1024
  gemm_nt_wmma_bf16<u16><<<dim3(8, 512, 1), 256, 0, stream>>>(
      xb, wkvb, kvb, BATCH * SEQN, 2 * INNER, DIM, DIM, DIM, 2 * INNER,
      0LL, 0LL, 0LL, 0LL, 1, 1.0f);
  reshape_kv<<<4096, 256, 0, stream>>>(kvb, kh, vT);

  // q = latents @ Wq^T, fused scale DH^-0.5 = 0.125 : M=512 N=1024 K=1024
  gemm_nt_wmma_bf16<u16><<<dim3(4, 8, 1), 256, 0, stream>>>(
      latb, wqb, qb, BATCH * SEQM, INNER, DIM, DIM, DIM, INNER,
      0LL, 0LL, 0LL, 0LL, 1, 0.125f);
  reshape_q<<<512, 256, 0, stream>>>(qb, qh);

  // sim[z] = qh[z] @ kh[z]^T : z=128, M=64 N=4096 K=64
  gemm_nt_wmma_bf16<float><<<dim3(16, 1, ZB), 256, 0, stream>>>(
      qh, kh, sim, SEQM, SEQN, DHEAD, DHEAD, DHEAD, SEQN,
      (long long)SEQM * DHEAD, (long long)SEQN * DHEAD,
      (long long)SEQM * SEQN, 0LL, 1, 1.0f);

  rowmax_kernel<<<ZB * SEQM, 256, 0, stream>>>(sim, mask, rmax);
  colvar_loss_kernel<<<dim3(SEQN / 256, ZB, 1), 256, 0, stream>>>(sim, mask, rmax, lacc);
  softmax_kernel<<<ZB * SEQM, 256, 0, stream>>>(sim, mask, rmax, attn);

  // outh[b,m,h*64+d] = attn[z] @ vT[z]^T : z=128, M=64 N=64 K=4096
  gemm_nt_wmma_bf16<u16><<<dim3(1, 1, ZB), 256, 0, stream>>>(
      attn, vT, outh, SEQM, DHEAD, SEQN, SEQN, SEQN, INNER,
      (long long)SEQM * SEQN, (long long)DHEAD * SEQN,
      (long long)SEQM * INNER, (long long)DHEAD, HEADS, 1.0f);

  // out = outh @ Wout^T : M=512 N=1024 K=1024 (f32 straight to d_out)
  gemm_nt_wmma_bf16<float><<<dim3(4, 8, 1), 256, 0, stream>>>(
      outh, woutb, out, BATCH * SEQM, DIM, INNER, INNER, INNER, DIM,
      0LL, 0LL, 0LL, 0LL, 1, 1.0f);

  loss_write_kernel<<<1, 1, 0, stream>>>(lacc, out + (long long)BATCH * SEQM * DIM);
}
